// HMM_62517543961019
// MI455X (gfx1250) — matrix operand order repaired and checked by reading, not compiled, not run on hardware
//
#include <hip/hip_runtime.h>
#include <hip/hip_bf16.h>
#include <math.h>

// ---------------------------------------------------------------------------
// Batched HMM forward (scaled linear-domain) for MI455X / gfx1250.
//   B=128, T=8192, S=65 (pad N->80, K->96), V=1024.
//   Per workgroup: 16 batch rows, 5 waves (one 16-col N tile each).
//   Transition matrix exp(log_trans) lives in VGPRs (3 x v16bf per wave) for
//   the entire kernel; per step: 3 chained v_wmma_f32_16x16x32_bf16.
//   One s_barrier per timestep (ping-pong bf16 state buffers in LDS).
// ---------------------------------------------------------------------------

#define S 65
#define VOCAB 1024
#define BATCH 128
#define TLEN 8192
#define ASTRIDE 104   // bf16 elems per state row: 208B, 16B aligned, bank-friendly

typedef __bf16 v8bf  __attribute__((ext_vector_type(8)));
typedef __bf16 v16bf __attribute__((ext_vector_type(16)));
typedef float  v8f   __attribute__((ext_vector_type(8)));

// Precompute exp(log_emit) into workspace (266 KB, stays L2-resident).
__global__ void exp_table_kernel(const float* __restrict__ in,
                                 float* __restrict__ out, int n) {
  int i = blockIdx.x * blockDim.x + threadIdx.x;
  if (i < n) out[i] = __expf(in[i]);
}

__launch_bounds__(160, 1)
__global__ void hmm_forward_kernel(const float* __restrict__ log_trans,
                                   const float* __restrict__ log_pi,
                                   const float* __restrict__ expEmit,
                                   const int*   __restrict__ obvs,
                                   float* __restrict__ out)
{
  __shared__ __align__(16) __bf16 sA[2][16 * ASTRIDE];   // ping-pong p-hat state

  const int tid  = threadIdx.x;
  const int lane = tid & 31;
  const int wid  = tid >> 5;       // N tile 0..4
  const int half = lane >> 4;      // wave half
  const int nn   = lane & 15;
  const int b0   = blockIdx.x * 16;
  const int j    = wid * 16 + nn;  // destination state column this lane produces

  // ---- init state buffers: p_hat0[r][k] = exp(log_pi[k]) ----
  for (int idx = tid; idx < 16 * ASTRIDE; idx += 160) {
    int k = idx % ASTRIDE;
    sA[0][idx] = (__bf16)((k < S) ? __expf(log_pi[k]) : 0.0f);
    sA[1][idx] = (__bf16)0.0f;
  }

  // ---- resident B registers: exp(log_trans) in bf16 WMMA B layout.
  // B (32x16 per chunk c): lanes 0-15 hold K=32c+{0-7,16-23}, lanes 16-31 hold
  // K=32c+{8-15,24-31}; matches A's K order; column = lane%16.
  v16bf bReg[3];
  #pragma unroll
  for (int c = 0; c < 3; ++c)
    #pragma unroll
    for (int i = 0; i < 16; ++i) {
      int k = 32 * c + 8 * half + (i & 7) + ((i >> 3) << 4);
      bReg[c][i] = (__bf16)((j < S && k < S) ? __expf(log_trans[k * S + j]) : 0.0f);
    }
  __syncthreads();

  // ---- software pipeline: obs fetched 2 steps ahead, emissions 1 step ahead
  int   oNxt[8];
  float eCur[8];
  #pragma unroll
  for (int v = 0; v < 8; ++v) {
    int r = 8 * half + v;                                   // D-row this value feeds
    int o0 = obvs[(b0 + r) * TLEN + 0];
    oNxt[v] = obvs[(b0 + r) * TLEN + 1];
    eCur[v] = (j < S) ? expEmit[j * VOCAB + o0] : 0.0f;
  }

  float logacc = 0.0f;

  for (int t = 0; t < TLEN; ++t) {
    const __bf16* Ar = sA[t & 1];
    __bf16*       Aw = sA[(t + 1) & 1];

    // --- load A (16x32 bf16 layout: lane = row nn; half 0 -> K {0-7,16-23},
    //     half 1 -> K {8-15,24-31}) and accumulate this lane's row-sum half.
    v16bf aReg[3];
    float rs0 = 0.0f, rs1 = 0.0f;
    #pragma unroll
    for (int c = 0; c < 3; ++c) {
      int kLo = 32 * c + 8 * half;
      v8bf lo = *(const v8bf*)&Ar[nn * ASTRIDE + kLo];
      v8bf hi = *(const v8bf*)&Ar[nn * ASTRIDE + kLo + 16];
      #pragma unroll
      for (int i = 0; i < 8; ++i) {
        aReg[c][i]     = lo[i];
        aReg[c][8 + i] = hi[i];
        rs0 += (float)lo[i];
        rs1 += (float)hi[i];
      }
    }
    // full row sum: lane^16 holds the complementary K half of the same row
    float rs = rs0 + rs1;
    float rowSum = rs + __shfl_xor(rs, 16, 32);
    float rcpOwn = __builtin_amdgcn_rcpf(rowSum);   // scale for row nn
    logacc += __logf(rowSum);                        // exact log of applied scale

    // --- D = A x Texp, f32 accumulate (3 chained WMMAs, K = 96) ---
    v8f d = {0.f, 0.f, 0.f, 0.f, 0.f, 0.f, 0.f, 0.f};
    d = __builtin_amdgcn_wmma_f32_16x16x32_bf16(false, aReg[0], false, bReg[0],
                                                (short)0, d, false, false);
    d = __builtin_amdgcn_wmma_f32_16x16x32_bf16(false, aReg[1], false, bReg[1],
                                                (short)0, d, false, false);
    d = __builtin_amdgcn_wmma_f32_16x16x32_bf16(false, aReg[2], false, bReg[2],
                                                (short)0, d, false, false);

    // --- prefetch emissions for t+1 (obs already in flight) and obs for t+2
    float eNxt[8];
    #pragma unroll
    for (int v = 0; v < 8; ++v)
      eNxt[v] = (j < S) ? expEmit[j * VOCAB + oNxt[v]] : 0.0f;
    int tn = (t + 2 < TLEN) ? t + 2 : TLEN - 1;
    int oN2[8];
    #pragma unroll
    for (int v = 0; v < 8; ++v)
      oN2[v] = obvs[(b0 + 8 * half + v) * TLEN + tn];

    // --- p'[r][j] = D[r][j] * (1/rowsum[r]) * e_t[r][j]; store bf16 ---
    #pragma unroll
    for (int v = 0; v < 8; ++v) {
      int r = 8 * half + v;
      float scale = __shfl(rcpOwn, r, 32);          // rcp lives in lane r
      float val = d[v] * scale * eCur[v];           // e==0 zeroes pad cols j>=65
      Aw[r * ASTRIDE + j] = (__bf16)val;
    }
    #pragma unroll
    for (int v = 0; v < 8; ++v) { eCur[v] = eNxt[v]; oNxt[v] = oN2[v]; }
    __syncthreads();
  }

  // ---- termination in LOG domain (exp(log_trans[:,0]) underflows f32):
  //      out[b] = logacc[b] + logsumexp_i( log p_T[b,i] + log_trans[i,0] )
  if (wid == 0) {
    const __bf16* Af = sA[TLEN & 1];
    int r  = nn;
    int i0 = half ? 33 : 0;
    int i1 = half ? S  : 33;
    float m = -3.0e38f;
    for (int i = i0; i < i1; ++i) {
      float p = (float)Af[r * ASTRIDE + i];
      float v = ((p > 0.0f) ? __logf(p) : -3.0e38f) + log_trans[i * S];
      m = fmaxf(m, v);
    }
    m = fmaxf(m, __shfl_xor(m, 16, 32));            // finite: max p >= rowsum/65
    float ssum = 0.0f;
    for (int i = i0; i < i1; ++i) {
      float p = (float)Af[r * ASTRIDE + i];
      if (p > 0.0f) ssum += __expf(__logf(p) + log_trans[i * S] - m);
    }
    ssum += __shfl_xor(ssum, 16, 32);
    if (half == 0) out[b0 + r] = logacc + m + __logf(ssum);
  }
}

extern "C" void kernel_launch(void* const* d_in, const int* in_sizes, int n_in,
                              void* d_out, int out_size, void* d_ws, size_t ws_size,
                              hipStream_t stream) {
  (void)in_sizes; (void)n_in; (void)out_size; (void)ws_size;
  const float* log_trans = (const float*)d_in[0];   // [65,65] f32
  const float* log_emit  = (const float*)d_in[1];   // [65,1024] f32
  const float* log_pi    = (const float*)d_in[2];   // [65] f32
  const int*   obvs      = (const int*)d_in[3];     // [128,8192] i32
  float*       out       = (float*)d_out;           // [128] f32
  float*       expEmit   = (float*)d_ws;            // 65*1024 f32 scratch

  int n = S * VOCAB;
  exp_table_kernel<<<(n + 255) / 256, 256, 0, stream>>>(log_emit, expEmit, n);
  hmm_forward_kernel<<<BATCH / 16, 160, 0, stream>>>(log_trans, log_pi,
                                                     expEmit, obvs, out);
}
